// Attention_74088185856351
// MI455X (gfx1250) — matrix operand order repaired and checked by reading, not compiled, not run on hardware
//
#include <hip/hip_runtime.h>
#include <hip/hip_bf16.h>

// ---------------------------------------------------------------------------
// Attention block for MI455X (gfx1250, wave32, WMMA).
//   B=8, N=1024, C=768, H=12, D=64.  ~65 GFLOP, AI >> HBM knee -> WMMA-bound.
//   All matmuls use v_wmma_f32_16x16x32_bf16 (bf16 in, f32 accumulate).
// ---------------------------------------------------------------------------

#define B_  8
#define N_  1024
#define C_  768
#define H_  12
#define D_  64
#define C3_ (3 * C_)
#define SCALE_ 0.125f   // D^-0.5 = 1/8

typedef __attribute__((ext_vector_type(16))) __bf16 v16bf;
typedef __attribute__((ext_vector_type(8)))  float  v8f;

// A-fragment (16x32 bf16, M x K) K-base for VGPR slot i, lane-half `half`
// per CDNA5 ISA 7.12.2: VGPR0..3 -> K={0..7}+8*half, VGPR4..7 -> K=16..23 / 24..31
__device__ __forceinline__ int kbase_a(int i, int half) {
    return ((i < 4) ? (2 * i) : (16 + 2 * (i - 4))) + 8 * half;
}
// B-fragment (32x16 bf16, K x N) K-base: same K order as the A-fragment
__device__ __forceinline__ int kbase_b(int i, int half) {
    return ((i < 4) ? (2 * i) : (16 + 2 * (i - 4))) + 8 * half;
}

// ---------------------------------------------------------------------------
// Kernel 1: fp32 -> bf16 conversion
// ---------------------------------------------------------------------------
__global__ void __launch_bounds__(256)
cvt_f32_bf16(const float* __restrict__ src, __bf16* __restrict__ dst, int n) {
    int i = blockIdx.x * blockDim.x + threadIdx.x;
    if (i < n) dst[i] = (__bf16)src[i];
}

// ---------------------------------------------------------------------------
// Kernel 2/5: GEMM  Cout[M][Ncol] = A[M][K] * Wt[Ncol][K]^T (+bias)
//   block = 256 threads = 8 waves; each wave owns a 32x64 tile:
//   per K-step: 2 A-fragments + 4 B-fragments -> 8 WMMAs (B reused 2x, A 4x).
//   Grid: (M/256, Ncol/64).
// ---------------------------------------------------------------------------
__global__ void __launch_bounds__(256)
gemm_bf16_wmma(const __bf16* __restrict__ A,
               const __bf16* __restrict__ Wt,
               const float* __restrict__ bias,
               float* __restrict__ Cout,
               int M, int Ncol, int K) {
    const int lane = threadIdx.x & 31;
    const int wave = threadIdx.x >> 5;
    const int half = lane >> 4;
    const int m16  = lane & 15;
    const int row0 = (blockIdx.x * 8 + wave) * 32;
    const int col0 = blockIdx.y * 64;

    v8f acc[2][4] = {};

    for (int kb = 0; kb < K; kb += 32) {
        v16bf a0, a1;
#pragma unroll
        for (int i = 0; i < 8; ++i) {
            const int kk = kbase_a(i, half);
            const __bf16* p0 = A + (size_t)(row0 + m16) * K + kb + kk;
            const __bf16* p1 = A + (size_t)(row0 + 16 + m16) * K + kb + kk;
            a0[2 * i] = p0[0]; a0[2 * i + 1] = p0[1];
            a1[2 * i] = p1[0]; a1[2 * i + 1] = p1[1];
        }
#pragma unroll
        for (int nt = 0; nt < 4; ++nt) {
            v16bf b;
#pragma unroll
            for (int i = 0; i < 8; ++i) {
                const int kk = kbase_b(i, half);
                const __bf16* bp = Wt + (size_t)(col0 + nt * 16 + m16) * K + kb + kk;
                b[2 * i]     = bp[0];
                b[2 * i + 1] = bp[1];
            }
            acc[0][nt] = __builtin_amdgcn_wmma_f32_16x16x32_bf16(
                false, a0, false, b, (short)0, acc[0][nt], false, false);
            acc[1][nt] = __builtin_amdgcn_wmma_f32_16x16x32_bf16(
                false, a1, false, b, (short)0, acc[1][nt], false, false);
        }
    }

#pragma unroll
    for (int mt = 0; mt < 2; ++mt)
#pragma unroll
        for (int nt = 0; nt < 4; ++nt) {
            const int col = col0 + nt * 16 + m16;
            const float bv = bias ? bias[col] : 0.0f;
#pragma unroll
            for (int r = 0; r < 8; ++r) {
                const int row = row0 + mt * 16 + r + 8 * half;  // C/D: vgpr r -> M=r+8*half
                Cout[(size_t)row * Ncol + col] = acc[mt][nt][r] + bv;
            }
        }
}

// ---------------------------------------------------------------------------
// Kernel 3: per-(b,h,n) standardization of q,k (unbiased, ddof=1) + bf16 pack
//   of q,k,v into [B*H, N, 64].  One wave per row; 32-lane shfl reduction.
// ---------------------------------------------------------------------------
__global__ void __launch_bounds__(256)
norm_pack(const float* __restrict__ qkv,   // [B, N, 2304]
          __bf16* __restrict__ qn, __bf16* __restrict__ kn, __bf16* __restrict__ vn) {
    const int lane = threadIdx.x & 31;
    const int wave = threadIdx.x >> 5;
    const int rid  = blockIdx.x * 8 + wave;          // (b*H + h)*N + n
    const int n = rid & (N_ - 1);
    const int h = (rid / N_) % H_;
    const int b = rid / (N_ * H_);

    const float* row = qkv + ((size_t)b * N_ + n) * C3_ + h * D_;

#pragma unroll
    for (int part = 0; part < 2; ++part) {           // 0 = q, 1 = k
        const float* p = row + part * C_;
        float x0 = p[lane], x1 = p[lane + 32];
        float s = x0 + x1;
#pragma unroll
        for (int m = 16; m >= 1; m >>= 1) s += __shfl_xor(s, m, 32);
        const float mu = s * (1.0f / 64.0f);
        const float d0 = x0 - mu, d1 = x1 - mu;
        float ss = d0 * d0 + d1 * d1;
#pragma unroll
        for (int m = 16; m >= 1; m >>= 1) ss += __shfl_xor(ss, m, 32);
        const float rs = rsqrtf(ss * (1.0f / 63.0f));   // unbiased (ddof=1)
        __bf16* dst = (part == 0 ? qn : kn) + (size_t)rid * D_;
        dst[lane]      = (__bf16)(d0 * rs);
        dst[lane + 32] = (__bf16)(d1 * rs);
    }
    // v: straight bf16 pack
    const float* pv = row + 2 * C_;
    vn[(size_t)rid * D_ + lane]      = (__bf16)pv[lane];
    vn[(size_t)rid * D_ + lane + 32] = (__bf16)pv[lane + 32];
}

// ---------------------------------------------------------------------------
// Kernel 4: flash-style attention per head.
//   Block = 128 threads = 4 waves; wave owns 16 query rows.  Per 32-key block:
//   * V tile (32x64) is loaded ONCE per block (coalesced b128) and stored
//     transposed into LDS (double-buffered) -> PV B-fragments become
//     two 16B ds_load_b128 reads per lane instead of 64 scalar global gathers
//     per wave (and global V traffic drops 4x).
//   * QK^T: 4 WMMA (K loads are contiguous -> global_load_b128, WGP$-shared).
//   * online softmax: 16-lane-group shfl reductions on the C-fragment rows.
//   * P restaged through per-wave LDS (s_wait_dscnt) into A-fragment layout.
//   * PV: 4 WMMA into a 16x64 f32 accumulator.
// ---------------------------------------------------------------------------
#define VST 40   // LDS row stride (shorts) for transposed V: 16B-aligned, bank-spread

__global__ void __launch_bounds__(128)
attn_wmma(const __bf16* __restrict__ Qn,   // [B*H, N, 64] normalized
          const __bf16* __restrict__ Kn,
          const __bf16* __restrict__ Vn,
          __bf16* __restrict__ Out) {      // [B, N, C] bf16
    __shared__ __bf16 ldsP[4][16 * 32];    // per-wave P staging (1 KB each)
    __shared__ __bf16 vt[2][D_ * VST];     // transposed V tile, double-buffered

    const int tid  = threadIdx.x;
    const int lane = tid & 31;
    const int wave = tid >> 5;
    const int half = lane >> 4;
    const int m16  = lane & 15;

    const int bh = blockIdx.x;             // 0..95
    const int b  = bh / H_;
    const int h  = bh % H_;
    const int q0 = blockIdx.y * 64 + wave * 16;

    const __bf16* Q = Qn + (size_t)bh * N_ * D_;
    const __bf16* K = Kn + (size_t)bh * N_ * D_;
    const __bf16* V = Vn + (size_t)bh * N_ * D_;

    // Q fragments for K-dim 0..31 and 32..63, kept resident all iterations
    v16bf aq[2];
#pragma unroll
    for (int t = 0; t < 2; ++t)
#pragma unroll
        for (int i = 0; i < 8; ++i) {
            const int kk = t * 32 + kbase_a(i, half);
            const __bf16* p = Q + (size_t)(q0 + m16) * D_ + kk;
            aq[t][2 * i]     = p[0];
            aq[t][2 * i + 1] = p[1];
        }

    v8f o[4] = {};
    float mr[8], lr[8];
#pragma unroll
    for (int r = 0; r < 8; ++r) { mr[r] = -1e30f; lr[r] = 0.0f; }

    __bf16* L = ldsP[wave];
    const int vrow  = tid >> 2;            // 0..31: key row within tile
    const int vdbas = (tid & 3) * 16;      // 0,16,32,48: d-chunk

    for (int key0 = 0; key0 < N_; key0 += 32) {
        const int buf = (key0 >> 5) & 1;

        // ---- cooperative transpose-load of V tile into LDS ----
        {
            const __bf16* src = V + (size_t)(key0 + vrow) * D_ + vdbas;
            __bf16 tmp[16];
#pragma unroll
            for (int j = 0; j < 16; ++j) tmp[j] = src[j];        // 2x b128 coalesced
#pragma unroll
            for (int j = 0; j < 16; ++j)
                vt[buf][(vdbas + j) * VST + vrow] = tmp[j];      // transposed scatter
        }
        __syncthreads();   // also orders reuse of buffer `buf` (double-buffered)

        // ---- S = Q * K^T for 32 keys: two 16x16 C tiles ----
        v8f s0 = {}, s1 = {};
#pragma unroll
        for (int t = 0; t < 2; ++t) {
            v16bf bk0, bk1;
#pragma unroll
            for (int i = 0; i < 8; ++i) {
                const int kk = t * 32 + kbase_b(i, half);
                const __bf16* p0 = K + (size_t)(key0 + m16) * D_ + kk;
                const __bf16* p1 = K + (size_t)(key0 + 16 + m16) * D_ + kk;
                bk0[2 * i] = p0[0]; bk0[2 * i + 1] = p0[1];
                bk1[2 * i] = p1[0]; bk1[2 * i + 1] = p1[1];
            }
            s0 = __builtin_amdgcn_wmma_f32_16x16x32_bf16(false, aq[t], false, bk0,
                                                         (short)0, s0, false, false);
            s1 = __builtin_amdgcn_wmma_f32_16x16x32_bf16(false, aq[t], false, bk1,
                                                         (short)0, s1, false, false);
        }

        // ---- online softmax over the 32 new columns ----
        float p0v[8], p1v[8], fsc[8];
#pragma unroll
        for (int r = 0; r < 8; ++r) {
            const float v0 = s0[r] * SCALE_;
            const float v1 = s1[r] * SCALE_;
            float rmx = fmaxf(v0, v1);
#pragma unroll
            for (int m = 8; m >= 1; m >>= 1) rmx = fmaxf(rmx, __shfl_xor(rmx, m, 32));
            const float mnew = fmaxf(mr[r], rmx);
            const float f    = __expf(mr[r] - mnew);
            p0v[r] = __expf(v0 - mnew);
            p1v[r] = __expf(v1 - mnew);
            float ps = p0v[r] + p1v[r];
#pragma unroll
            for (int m = 8; m >= 1; m >>= 1) ps += __shfl_xor(ps, m, 32);
            lr[r] = lr[r] * f + ps;
            mr[r] = mnew;
            fsc[r] = f;
        }
#pragma unroll
        for (int nt = 0; nt < 4; ++nt)
#pragma unroll
            for (int r = 0; r < 8; ++r) o[nt][r] *= fsc[r];

        // ---- restage P (C-layout) -> LDS row-major 16x32 -> A-fragment ----
#pragma unroll
        for (int r = 0; r < 8; ++r) {
            const int row = r + 8 * half;
            L[row * 32 + m16]      = (__bf16)p0v[r];
            L[row * 32 + 16 + m16] = (__bf16)p1v[r];
        }
        asm volatile("s_wait_dscnt 0" ::: "memory");   // CDNA5 split DS counter
        v16bf ap;
#pragma unroll
        for (int i = 0; i < 8; ++i) {
            const int kk = kbase_a(i, half);
            ap[2 * i]     = L[m16 * 32 + kk];
            ap[2 * i + 1] = L[m16 * 32 + kk + 1];
        }

        // ---- O += P * V  (K=32, four 16-wide N tiles), V from LDS ----
#pragma unroll
        for (int nt = 0; nt < 4; ++nt) {
            v16bf bv;
            const __bf16* vrowp = &vt[buf][(nt * 16 + m16) * VST];
#pragma unroll
            for (int i = 0; i < 8; ++i) {
                const int kk = kbase_b(i, half);       // two 8-short runs: k=8*half+0..7, 16+8*half+0..7
                bv[2 * i]     = vrowp[kk];
                bv[2 * i + 1] = vrowp[kk + 1];
            }
            o[nt] = __builtin_amdgcn_wmma_f32_16x16x32_bf16(false, ap, false, bv,
                                                            (short)0, o[nt], false, false);
        }
    }

    // ---- epilogue: normalize by row sums, write [B,N,C] bf16 ----
    float inv[8];
#pragma unroll
    for (int r = 0; r < 8; ++r) inv[r] = 1.0f / lr[r];
#pragma unroll
    for (int nt = 0; nt < 4; ++nt) {
        const int c = h * D_ + nt * 16 + m16;
#pragma unroll
        for (int r = 0; r < 8; ++r) {
            const int row = q0 + r + 8 * half;
            Out[((size_t)b * N_ + row) * C_ + c] = (__bf16)(o[nt][r] * inv[r]);
        }
    }
}

// ---------------------------------------------------------------------------
// Host-side launcher
// ---------------------------------------------------------------------------
extern "C" void kernel_launch(void* const* d_in, const int* in_sizes, int n_in,
                              void* d_out, int out_size, void* d_ws, size_t ws_size,
                              hipStream_t stream) {
    const float* x      = (const float*)d_in[0];   // [B,N,C]
    const float* qkv_w  = (const float*)d_in[1];   // [3C,C]
    const float* proj_w = (const float*)d_in[2];   // [C,C]
    const float* proj_b = (const float*)d_in[3];   // [C]
    float* out = (float*)d_out;                    // [B,N,C] f32

    char* ws = (char*)d_ws;
    size_t off = 0;
    auto carve = [&](size_t bytes) -> char* {
        char* p = ws + off;
        off += (bytes + 255) & ~(size_t)255;
        return p;
    };

    const size_t M  = (size_t)B_ * N_;                       // 8192
    __bf16* x_bf   = (__bf16*)carve(M * C_ * 2);             // 12.6 MB
    __bf16* wq_bf  = (__bf16*)carve((size_t)C3_ * C_ * 2);   //  3.5 MB
    __bf16* wp_bf  = (__bf16*)carve((size_t)C_ * C_ * 2);    //  1.2 MB
    float*  qkv_f  = (float*) carve(M * C3_ * 4);            // 75.5 MB
    __bf16* q_bf   = (__bf16*)carve(M * C_ * 2);             // 12.6 MB
    __bf16* k_bf   = (__bf16*)carve(M * C_ * 2);             // 12.6 MB
    __bf16* v_bf   = (__bf16*)carve(M * C_ * 2);             // 12.6 MB
    __bf16* att_bf = (__bf16*)qkv_f;                         // reuse dead f32 region

    // 1) bf16 conversions
    {
        int n1 = (int)(M * C_);
        cvt_f32_bf16<<<(n1 + 255) / 256, 256, 0, stream>>>(x, x_bf, n1);
        int n2 = C3_ * C_;
        cvt_f32_bf16<<<(n2 + 255) / 256, 256, 0, stream>>>(qkv_w, wq_bf, n2);
        int n3 = C_ * C_;
        cvt_f32_bf16<<<(n3 + 255) / 256, 256, 0, stream>>>(proj_w, wp_bf, n3);
    }

    // 2) QKV GEMM: [8192,768] x [2304,768]^T -> [8192,2304] f32
    gemm_bf16_wmma<<<dim3((int)M / 256, C3_ / 64), 256, 0, stream>>>(
        x_bf, wq_bf, nullptr, qkv_f, (int)M, C3_, C_);

    // 3) normalize q,k (ddof=1) + pack q,k,v -> [B*H, N, 64] bf16
    norm_pack<<<(B_ * H_ * N_) / 8, 256, 0, stream>>>(qkv_f, q_bf, k_bf, v_bf);

    // 4) attention -> [B,N,C] bf16
    attn_wmma<<<dim3(B_ * H_, N_ / 64), 128, 0, stream>>>(q_bf, k_bf, v_bf, att_bf);

    // 5) projection GEMM + bias -> f32 output
    gemm_bf16_wmma<<<dim3((int)M / 256, C_ / 64), 256, 0, stream>>>(
        att_bf, wp_bf, proj_b, out, (int)M, C_, C_);
}
